// Model_39032662786617
// MI455X (gfx1250) — matrix operand order repaired and checked by reading, not compiled, not run on hardware
//
#include <hip/hip_runtime.h>
#include <hip/hip_bf16.h>
#include <math.h>

#define BATCH 8
#define H 256
#define W 256
#define HW (H*W)
#define NPIX (BATCH*HW)

typedef _Float16 v16h __attribute__((ext_vector_type(16)));
typedef _Float16 v8h  __attribute__((ext_vector_type(8)));
typedef float    v8f  __attribute__((ext_vector_type(8)));

// ---------------------------------------------------------------- k_diff
__global__ void __launch_bounds__(256) k_diff(const float* __restrict__ o,
                                              const float* __restrict__ r,
                                              float* __restrict__ d) {
    int i = blockIdx.x * 256 + threadIdx.x;
    d[i] = o[i] - r[i];
}

// ------------------------------------------------------- conv1: 1 -> 16
__global__ void __launch_bounds__(256) k_conv1(const float* __restrict__ in,
                                               const float* __restrict__ w,
                                               const float* __restrict__ bias,
                                               _Float16* __restrict__ h1) {
    int g = blockIdx.x * 256 + threadIdx.x;
    int b = g >> 16; int pix = g & (HW - 1);
    int y = pix >> 8, x = pix & 255;
    const float* img = in + (size_t)b * HW;
    float v[9];
#pragma unroll
    for (int t = 0; t < 9; t++) {
        int dy = t / 3 - 1, dx = t % 3 - 1;
        int yy = y + dy, xx = x + dx;
        v[t] = (yy >= 0 && yy < H && xx >= 0 && xx < W) ? img[yy * W + xx] : 0.f;
    }
    _Float16* o = h1 + (size_t)g * 16;
#pragma unroll
    for (int c = 0; c < 16; c++) {
        float a = bias[c];
#pragma unroll
        for (int t = 0; t < 9; t++) a += w[c * 9 + t] * v[t];
        o[c] = (_Float16)fmaxf(a, 0.f);
    }
}

// ------------------------------------------- conv2: 16 -> 64 (WMMA GEMM)
// K = 9*16 = 144, padded to 160; k-order = tap*16 + cin
#define K2PAD 168
__global__ void __launch_bounds__(256) k_conv2(const _Float16* __restrict__ h1,
                                               const float* __restrict__ w,
                                               const float* __restrict__ bias,
                                               _Float16* __restrict__ h2) {
    __shared__ _Float16 Al[128 * K2PAD];
    __shared__ _Float16 Wl[64 * K2PAD];
    int tid = threadIdx.x;
    int base = blockIdx.x * 128;            // global pixel base (row segment)
    int b = base >> 16; int pix0 = base & (HW - 1);
    int y = pix0 >> 8, x0 = pix0 & 255;

    // weights -> Wl[n][k]  (B matrix stored N-major so B frags are contiguous)
    for (int idx = tid; idx < 64 * K2PAD; idx += 256) {
        int n = idx / K2PAD, k = idx % K2PAD;
        float v = 0.f;
        if (k < 144) { int tap = k >> 4, c = k & 15; v = w[(n * 16 + c) * 9 + tap]; }
        Wl[idx] = (_Float16)v;
    }
    // A tile: Al[p][k] = h1[neigh(p,tap)][cin]
    const _Float16* h1b = h1 + (size_t)b * HW * 16;
    for (int idx = tid; idx < 128 * K2PAD; idx += 256) {
        int p = idx & 127; int k = idx >> 7;
        _Float16 hv = (_Float16)0.f;
        if (k < 144) {
            int tap = k >> 4, c = k & 15;
            int dy = tap / 3 - 1, dx = tap % 3 - 1;
            int yy = y + dy, xx = x0 + p + dx;
            if (yy >= 0 && yy < H && xx >= 0 && xx < W)
                hv = h1b[((size_t)(yy * W + xx)) * 16 + c];
        }
        Al[p * K2PAD + k] = hv;
    }
    __syncthreads();

    int wv = tid >> 5, lane = tid & 31, m = lane & 15, hi = lane >> 4;
    int mb = wv * 16;
    v8f acc[4] = {};
#pragma unroll
    for (int kb = 0; kb < 160; kb += 32) {
        v8h alo = *(const v8h*)&Al[(mb + m) * K2PAD + kb + hi * 16];
        v8h ahi = *(const v8h*)&Al[(mb + m) * K2PAD + kb + hi * 16 + 8];
        v16h a;
#pragma unroll
        for (int i = 0; i < 8; i++) { a[i] = alo[i]; a[i + 8] = ahi[i]; }
#pragma unroll
        for (int nt = 0; nt < 4; nt++) {
            v16h bf = *(const v16h*)&Wl[(nt * 16 + m) * K2PAD + kb + hi * 16];
            acc[nt] = __builtin_amdgcn_wmma_f32_16x16x32_f16(
                false, a, false, bf, (short)0, acc[nt], false, false);
        }
    }
    // epilogue: bias + relu -> h2[pix][64] f16
    _Float16* h2b = h2 + (size_t)base * 64;
#pragma unroll
    for (int nt = 0; nt < 4; nt++) {
        int c = nt * 16 + m;
        float bc = bias[c];
#pragma unroll
        for (int r = 0; r < 8; r++) {
            int p = mb + r + 8 * hi;
            float v = acc[nt][r] + bc;
            h2b[(size_t)p * 64 + c] = (_Float16)fmaxf(v, 0.f);
        }
    }
}

// -------- conv3 (64->81, WMMA K=576) + softmax(81) + dynamic 9x9 apply
#define K3AP 72
__global__ void __launch_bounds__(256) k_conv3_apply(const _Float16* __restrict__ h2,
                                                     const float* __restrict__ w3,
                                                     const float* __restrict__ b3,
                                                     const float* __restrict__ xin,
                                                     float* __restrict__ outp) {
    __shared__ _Float16 Al[128 * K3AP];
    __shared__ _Float16 Wl[96 * K3AP];
    __shared__ float xl[9 * 136];
    int tid = threadIdx.x;
    int basep = blockIdx.x * 128;
    int b = basep >> 16; int pix0 = basep & (HW - 1);
    int y = pix0 >> 8, x0 = pix0 & 255;

    // stage raw-input patch rows y-4..y+4, cols x0-4..x0+131 (zero padded)
    const float* img = xin + (size_t)b * HW;
    for (int idx = tid; idx < 9 * 136; idx += 256) {
        int rr = idx / 136, cc = idx % 136;
        int yy = y + rr - 4, xx = x0 + cc - 4;
        xl[idx] = (yy >= 0 && yy < H && xx >= 0 && xx < W) ? img[yy * W + xx] : 0.f;
    }

    int wv = tid >> 5, lane = tid & 31, m = lane & 15, hi = lane >> 4;
    int mb = wv * 16;
    v8f acc[6] = {};
    const _Float16* h2b = h2 + (size_t)b * HW * 64;

    for (int tap = 0; tap < 9; tap++) {
        int dy = tap / 3 - 1, dx = tap % 3 - 1;
        __syncthreads();                    // protect previous iteration reads
        // B tile for this tap: Wl[n][c] = w3[n][c][tap] (n>=81 -> 0)
        for (int idx = tid; idx < 96 * 64; idx += 256) {
            int n = idx >> 6, c = idx & 63;
            float v = (n < 81) ? w3[((size_t)n * 64 + c) * 9 + tap] : 0.f;
            Wl[n * K3AP + c] = (_Float16)v;
        }
        // A tile: Al[p][c] = h2[neigh][c]  (contiguous 64-half row copies)
        int yy = y + dy;
        for (int idx = tid; idx < 128 * 64; idx += 256) {
            int p = idx >> 6, c = idx & 63;
            int xx = x0 + p + dx;
            _Float16 v = (_Float16)0.f;
            if (yy >= 0 && yy < H && xx >= 0 && xx < W)
                v = h2b[((size_t)(yy * W + xx)) * 64 + c];
            Al[p * K3AP + c] = v;
        }
        __syncthreads();
#pragma unroll
        for (int kb = 0; kb < 64; kb += 32) {
            v8h alo = *(const v8h*)&Al[(mb + m) * K3AP + kb + hi * 16];
            v8h ahi = *(const v8h*)&Al[(mb + m) * K3AP + kb + hi * 16 + 8];
            v16h a;
#pragma unroll
            for (int i = 0; i < 8; i++) { a[i] = alo[i]; a[i + 8] = ahi[i]; }
#pragma unroll
            for (int nt = 0; nt < 6; nt++) {
                v16h bf = *(const v16h*)&Wl[(nt * 16 + m) * K3AP + kb + hi * 16];
                acc[nt] = __builtin_amdgcn_wmma_f32_16x16x32_f16(
                    false, a, false, bf, (short)0, acc[nt], false, false);
            }
        }
    }

    // per-lane channel metadata
    int   nn[6]; float bb[6];
#pragma unroll
    for (int nt = 0; nt < 6; nt++) {
        int n = nt * 16 + m;
        nn[nt] = n;
        bb[nt] = (n < 81) ? b3[n] : 0.f;
    }
    // softmax over 81 (across the 16 lanes of each half-group + 6 tiles),
    // then dynamic 9x9 filter application against xl
#pragma unroll
    for (int r = 0; r < 8; r++) {
        int ploc = mb + r + 8 * hi;          // pixel within 128-segment
        float v[6]; float mx = -3.0e38f;
#pragma unroll
        for (int nt = 0; nt < 6; nt++) {
            v[nt] = (nn[nt] < 81) ? (acc[nt][r] + bb[nt]) : -3.0e38f;
            mx = fmaxf(mx, v[nt]);
        }
        mx = fmaxf(mx, __shfl_xor(mx, 1, 32));
        mx = fmaxf(mx, __shfl_xor(mx, 2, 32));
        mx = fmaxf(mx, __shfl_xor(mx, 4, 32));
        mx = fmaxf(mx, __shfl_xor(mx, 8, 32));
        float e[6], sum = 0.f;
#pragma unroll
        for (int nt = 0; nt < 6; nt++) {
            e[nt] = (nn[nt] < 81) ? __expf(v[nt] - mx) : 0.f;
            sum += e[nt];
        }
        sum += __shfl_xor(sum, 1, 32);
        sum += __shfl_xor(sum, 2, 32);
        sum += __shfl_xor(sum, 4, 32);
        sum += __shfl_xor(sum, 8, 32);
        float inv = 1.f / sum;
        float part = 0.f;
#pragma unroll
        for (int nt = 0; nt < 6; nt++) {
            if (nn[nt] < 81) {
                int i = nn[nt] / 9, j = nn[nt] % 9;    // tap (dy=i-4, dx=j-4)
                part += e[nt] * inv * xl[i * 136 + ploc + j];
            }
        }
        part += __shfl_xor(part, 1, 32);
        part += __shfl_xor(part, 2, 32);
        part += __shfl_xor(part, 4, 32);
        part += __shfl_xor(part, 8, 32);
        if (m == 0) outp[(size_t)basep + ploc] = part;
    }
}

// ------------------------------------------------------- c1: 2 -> 64 relu
__global__ void __launch_bounds__(256) k_c1(const float* __restrict__ xr,
                                            const float* __restrict__ gr,
                                            const float* __restrict__ w,
                                            const float* __restrict__ bias,
                                            _Float16* __restrict__ hbuf) {
    int g = blockIdx.x * 256 + threadIdx.x;
    int b = g >> 16; int pix = g & (HW - 1);
    int y = pix >> 8, x = pix & 255;
    const float* x0 = xr + (size_t)b * HW;
    const float* g0 = gr + (size_t)b * HW;
    float v[18];
#pragma unroll
    for (int t = 0; t < 9; t++) {
        int dy = t / 3 - 1, dx = t % 3 - 1;
        int yy = y + dy, xx = x + dx;
        bool in = (yy >= 0 && yy < H && xx >= 0 && xx < W);
        v[t]     = in ? x0[yy * W + xx] : 0.f;
        v[9 + t] = in ? g0[yy * W + xx] : 0.f;
    }
    _Float16* o = hbuf + (size_t)g * 64;
    for (int c = 0; c < 64; c++) {
        float a = bias[c];
#pragma unroll
        for (int t = 0; t < 18; t++) a += w[c * 18 + t] * v[t];
        o[c] = (_Float16)fmaxf(a, 0.f);
    }
}

// ------------------------------------ c2: 64 -> 1, then out = s + d
__global__ void __launch_bounds__(256) k_c2(const _Float16* __restrict__ hbuf,
                                            const float* __restrict__ w,
                                            const float* __restrict__ bias,
                                            const float* __restrict__ dbuf,
                                            float* __restrict__ out_out,
                                            float* __restrict__ out_s) {
    int g = blockIdx.x * 256 + threadIdx.x;
    int b = g >> 16; int pix = g & (HW - 1);
    int y = pix >> 8, x = pix & 255;
    float acc = bias[0];
#pragma unroll
    for (int t = 0; t < 9; t++) {
        int dy = t / 3 - 1, dx = t % 3 - 1;
        int yy = y + dy, xx = x + dx;
        if (yy >= 0 && yy < H && xx >= 0 && xx < W) {
            const v8h* hp = (const v8h*)(hbuf + ((size_t)(b * HW + yy * W + xx)) * 64);
#pragma unroll
            for (int cv = 0; cv < 8; cv++) {
                v8h hh = hp[cv];
#pragma unroll
                for (int e = 0; e < 8; e++)
                    acc += (float)hh[e] * w[(cv * 8 + e) * 9 + t];
            }
        }
    }
    float dv = dbuf[g];
    out_s[g]   = acc;
    out_out[g] = acc + dv;
}

// ----------------------------------------------------------------------
extern "C" void kernel_launch(void* const* d_in, const int* in_sizes, int n_in,
                              void* d_out, int out_size, void* d_ws, size_t ws_size,
                              hipStream_t stream) {
    (void)in_sizes; (void)n_in; (void)out_size; (void)ws_size;
    const float* origin   = (const float*)d_in[0];
    const float* residual = (const float*)d_in[1];
    const float* fx_w1 = (const float*)d_in[2];  const float* fx_b1 = (const float*)d_in[3];
    const float* fx_w2 = (const float*)d_in[4];  const float* fx_b2 = (const float*)d_in[5];
    const float* fx_w3 = (const float*)d_in[6];  const float* fx_b3 = (const float*)d_in[7];
    const float* fg_w1 = (const float*)d_in[8];  const float* fg_b1 = (const float*)d_in[9];
    const float* fg_w2 = (const float*)d_in[10]; const float* fg_b2 = (const float*)d_in[11];
    const float* fg_w3 = (const float*)d_in[12]; const float* fg_b3 = (const float*)d_in[13];
    const float* c1_w  = (const float*)d_in[14]; const float* c1_b  = (const float*)d_in[15];
    const float* c2_w  = (const float*)d_in[16]; const float* c2_b  = (const float*)d_in[17];

    float* out_out = (float*)d_out;                 // [0, NPIX)
    float* out_s   = (float*)d_out + (size_t)NPIX;  // [NPIX, 2*NPIX)
    float* dbuf    = (float*)d_out + (size_t)2 * NPIX;  // d lives in output slot 2

    char* ws = (char*)d_ws;
    _Float16* h1  = (_Float16*)ws;                          // 16 MB  [pix][16]
    _Float16* h2  = (_Float16*)(ws + ((size_t)16 << 20));   // 64 MB  [pix][64]
    float*    xb  = (float*)(ws + ((size_t)80 << 20));      // 2 MB   x_
    float*    gb  = (float*)(ws + ((size_t)82 << 20));      // 2 MB   g_

    dim3 blk(256);
    // d = origin - residual  (also the g input and final residual add)
    k_diff<<<NPIX / 256, blk, 0, stream>>>(origin, residual, dbuf);

    // branch X  (input = residual)
    k_conv1<<<NPIX / 256, blk, 0, stream>>>(residual, fx_w1, fx_b1, h1);
    k_conv2<<<NPIX / 128, blk, 0, stream>>>(h1, fx_w2, fx_b2, h2);
    k_conv3_apply<<<NPIX / 128, blk, 0, stream>>>(h2, fx_w3, fx_b3, residual, xb);

    // branch G  (input = d)
    k_conv1<<<NPIX / 256, blk, 0, stream>>>(dbuf, fg_w1, fg_b1, h1);
    k_conv2<<<NPIX / 128, blk, 0, stream>>>(h1, fg_w2, fg_b2, h2);
    k_conv3_apply<<<NPIX / 128, blk, 0, stream>>>(h2, fg_w3, fg_b3, dbuf, gb);

    // fusion tail (h2 buffer reused as h)
    k_c1<<<NPIX / 256, blk, 0, stream>>>(xb, gb, c1_w, c1_b, h2);
    k_c2<<<NPIX / 256, blk, 0, stream>>>(h2, c2_w, c2_b, dbuf, out_out, out_s);
}
